// SFFM_40742059770214
// MI455X (gfx1250) — matrix operand order repaired and checked by reading, not compiled, not run on hardware
//
#include <hip/hip_runtime.h>
#include <hip/hip_bf16.h>
#include <math.h>

// ---------------------------------------------------------------------------
// SFFM (FFA recurrent feature model) for MI455X / gfx1250, wave32 + WMMA bf16.
// GEMMs via v_wmma_f32_16x16x32_bf16 (f32 accumulate). B tiles staged to LDS
// with gfx1250 async-to-LDS copies (double buffered); sequential complex scan
// parallelized as a 3-phase chunked associative scan.
// ---------------------------------------------------------------------------

typedef __bf16 bf16_t;
typedef __attribute__((ext_vector_type(16))) __bf16 v16bf;
typedef __attribute__((ext_vector_type(8)))  __bf16 v8bf;
typedef __attribute__((ext_vector_type(4)))  __bf16 v4bf;
typedef __attribute__((ext_vector_type(8)))  float  v8f;
typedef __attribute__((ext_vector_type(4)))  int    v4i;

#define T_DIM   16384
#define IN_DIM  1024
#define TRACE   64
#define CTX     16
#define OUT_DIM 1024
#define NCH     (TRACE * CTX)      // 1024 scan channels
#define KMIX    (3 * NCH)          // 3072 feature columns
#define NCHUNK  256
#define CLEN    (T_DIM / NCHUNK)   // 64 steps per chunk

// z-kernel tiling: 4 waves/block, each wave 16Mx64N; block = 64Mx64N.
#define ZWAVES  4
#define BSTRIDE 48                  // bytes per (ntile,lane) B segment in LDS
#define SBUF    (4 * 32 * BSTRIDE)  // one B stage: 4 ntiles x 32 lanes = 6 KB

// ---- gfx1250 async-to-LDS support (guarded; sync fallback keeps compiling) --
#if defined(__has_builtin)
# if __has_builtin(__builtin_amdgcn_global_load_async_to_lds_b128)
#  define SFFM_HAVE_ASYNC_LDS 1
# endif
# if __has_builtin(__builtin_amdgcn_s_wait_asynccnt)
#  define SFFM_HAVE_WAIT_ASYNC 1
# endif
#endif

// Builtin signature (from clang diagnostic): (int4 AS1*, int4 AS3*, imm, imm)
typedef __attribute__((address_space(1))) v4i* g1v4ip;
typedef __attribute__((address_space(3))) v4i* l3v4ip;

// Copy one 32-byte B segment global->LDS (async on gfx1250 toolchains).
static __device__ inline void sffm_cp32(const bf16_t* g, char* l) {
#if defined(SFFM_HAVE_ASYNC_LDS)
  g1v4ip gp = (g1v4ip)const_cast<bf16_t*>(g);
  l3v4ip lp = (l3v4ip)l;
  __builtin_amdgcn_global_load_async_to_lds_b128(gp, lp, 0, 0);
  __builtin_amdgcn_global_load_async_to_lds_b128(gp + 1, lp + 1, 0, 0);
#else
  *(v8bf*)l = *(const v8bf*)g;
  *(v8bf*)(l + 16) = *(const v8bf*)(g + 8);
#endif
}

static __device__ inline void sffm_wait_async() {
#if defined(SFFM_HAVE_ASYNC_LDS)
# if defined(SFFM_HAVE_WAIT_ASYNC)
  __builtin_amdgcn_s_wait_asynccnt(0);
# else
  asm volatile("s_wait_asynccnt 0x0" ::: "memory");
# endif
#endif
}

// Build a v16bf A/B operand from two contiguous 16-byte chunks.
static __device__ inline v16bf ld16x2(const bf16_t* p0, const bf16_t* p1) {
  v8bf lo = *(const v8bf*)p0;
  v8bf hi = *(const v8bf*)p1;
  return __builtin_shufflevector(lo, hi, 0,1,2,3,4,5,6,7,8,9,10,11,12,13,14,15);
}

// ---------------------------------------------------------------------------
// f32 -> bf16 conversion (4 elems / thread)
// ---------------------------------------------------------------------------
__global__ void sffm_cvt_bf16(const float* __restrict__ in,
                              bf16_t* __restrict__ out, int n4) {
  int i = blockIdx.x * blockDim.x + threadIdx.x;
  if (i < n4) {
    float4 v = ((const float4*)in)[i];
    v4bf o = { (bf16_t)v.x, (bf16_t)v.y, (bf16_t)v.z, (bf16_t)v.w };
    ((v4bf*)out)[i] = o;
  }
}

// ---------------------------------------------------------------------------
// pre = rownorm(x @ W_pre.T + bias_pre).  One wave per 16 rows; 16Mx64N tile,
// K=1024 via v_wmma_f32_16x16x32_bf16; bias + L2-norm fused (shfl_xor within
// the 16-lane halves of the wave32).
// ---------------------------------------------------------------------------
__global__ __launch_bounds__(32) void sffm_pre_kernel(
    const bf16_t* __restrict__ xb, const bf16_t* __restrict__ wb,
    const float* __restrict__ bias, float* __restrict__ pre) {
  const int lane = threadIdx.x;
  const int half = lane >> 4;
  const int l15  = lane & 15;
  const int m0   = blockIdx.x * 16;

  v8f acc[4] = {};
  const bf16_t* arow = xb + (size_t)(m0 + l15) * IN_DIM;
  v16bf acur = ld16x2(arow + half * 16, arow + half * 16 + 8);
  for (int k0 = 0; k0 < IN_DIM; k0 += 32) {
    v16bf anext = acur;
    if (k0 + 32 < IN_DIM)
      anext = ld16x2(arow + k0 + 32 + half * 16, arow + k0 + 40 + half * 16);
#pragma unroll
    for (int nt = 0; nt < 4; ++nt) {
      const bf16_t* brow = wb + (size_t)(nt * 16 + l15) * IN_DIM + k0 + half * 16;
      v16bf b = ld16x2(brow, brow + 8);
      acc[nt] = __builtin_amdgcn_wmma_f32_16x16x32_bf16(
          false, acur, false, b, (short)0, acc[nt], false, false);
    }
    acur = anext;
  }

  float bv[4];
#pragma unroll
  for (int nt = 0; nt < 4; ++nt) bv[nt] = bias[nt * 16 + l15];

#pragma unroll
  for (int i = 0; i < 8; ++i) {         // row = m0 + i + 8*half
    float s = 0.f;
#pragma unroll
    for (int nt = 0; nt < 4; ++nt) {
      float v = acc[nt][i] + bv[nt];
      acc[nt][i] = v;
      s += v * v;
    }
    s += __shfl_xor(s, 1, 32);
    s += __shfl_xor(s, 2, 32);
    s += __shfl_xor(s, 4, 32);
    s += __shfl_xor(s, 8, 32);          // stays inside the 16-lane half
    float inv = 1.0f / (1e-6f + sqrtf(s));
#pragma unroll
    for (int nt = 0; nt < 4; ++nt) acc[nt][i] *= inv;
  }

#pragma unroll
  for (int i = 0; i < 8; ++i) {
    int row = m0 + i + half * 8;
#pragma unroll
    for (int nt = 0; nt < 4; ++nt)
      pre[(size_t)row * TRACE + nt * 16 + l15] = acc[nt][i];
  }
}

// ---------------------------------------------------------------------------
// Scan helpers: gamma = exp(-|a[trace]|) * (cos b[ctx] + i sin b[ctx])
// ---------------------------------------------------------------------------
static __device__ inline void chan_gamma(int ch, const float* a_param,
                                         const float* b_param,
                                         float& gre, float& gim) {
  float mag = expf(-fabsf(a_param[ch >> 4]));
  float s, c;
  sincosf(b_param[ch & 15], &s, &c);
  gre = mag * c;
  gim = mag * s;
}

// Phase A: per (chunk, channel) local composition (A_prod, b_comb).
__global__ void sffm_scan_phaseA(const float* __restrict__ pre,
                                 const int* __restrict__ start,
                                 const float* __restrict__ a_param,
                                 const float* __restrict__ b_param,
                                 float4* __restrict__ chunkAB) {
  int ch = blockIdx.y * blockDim.x + threadIdx.x;
  int c  = blockIdx.x;
  float gre, gim;
  chan_gamma(ch, a_param, b_param, gre, gim);
  int tr = ch >> 4;
  float Are = 1.f, Aim = 0.f, bre = 0.f, bim = 0.f;
  int t0 = c * CLEN;
  for (int j = 0; j < CLEN; ++j) {
    int t = t0 + j;
    __builtin_prefetch(&pre[(size_t)(t + 16) * TRACE + tr], 0, 1);
    float p = pre[(size_t)t * TRACE + tr];
    if (start[t]) {
      Are = 0.f; Aim = 0.f; bre = p; bim = 0.f;
    } else {
      float nbre = gre * bre - gim * bim + p;
      float nbim = gre * bim + gim * bre;
      bre = nbre; bim = nbim;
      float nAre = gre * Are - gim * Aim;
      float nAim = gre * Aim + gim * Are;
      Are = nAre; Aim = nAim;
    }
  }
  chunkAB[(size_t)c * NCH + ch] = make_float4(Are, Aim, bre, bim);
}

// Phase B: serial prefix over 256 chunks per channel; emits carry-in state for
// each chunk and the final state (complex64 interleaved) to d_out tail.
__global__ __launch_bounds__(NCH) void sffm_scan_phaseB(
    const float4* __restrict__ chunkAB, const float* __restrict__ state0,
    float2* __restrict__ carry, float* __restrict__ final_out) {
  int ch = threadIdx.x;
  float sre = state0[ch], sim = 0.f;   // state cast to complex (im = 0)
  for (int c = 0; c < NCHUNK; ++c) {
    carry[(size_t)c * NCH + ch] = make_float2(sre, sim);
    float4 ab = chunkAB[(size_t)c * NCH + ch];
    float nre = ab.x * sre - ab.y * sim + ab.z;
    float nim = ab.x * sim + ab.y * sre + ab.w;
    sre = nre; sim = nim;
  }
  final_out[2 * ch]     = sre;
  final_out[2 * ch + 1] = sim;
}

// Phase C: replay chunk with carry-in, emit [log1p|s|, im/|s|, re/|s|] as the
// bf16 feature matrix (A-operand of the big GEMM).
__global__ void sffm_scan_phaseC(const float* __restrict__ pre,
                                 const int* __restrict__ start,
                                 const float* __restrict__ a_param,
                                 const float* __restrict__ b_param,
                                 const float2* __restrict__ carry,
                                 bf16_t* __restrict__ feats) {
  int ch = blockIdx.y * blockDim.x + threadIdx.x;
  int c  = blockIdx.x;
  float gre, gim;
  chan_gamma(ch, a_param, b_param, gre, gim);
  int tr = ch >> 4;
  float2 s0 = carry[(size_t)c * NCH + ch];
  float sre = s0.x, sim = s0.y;
  int t0 = c * CLEN;
  for (int j = 0; j < CLEN; ++j) {
    int t = t0 + j;
    __builtin_prefetch(&pre[(size_t)(t + 16) * TRACE + tr], 0, 1);
    float p = pre[(size_t)t * TRACE + tr];
    if (start[t]) {
      sre = p; sim = 0.f;
    } else {
      float nre = gre * sre - gim * sim + p;
      float nim = gre * sim + gim * sre;
      sre = nre; sim = nim;
    }
    float r   = sqrtf(sre * sre + sim * sim);
    float inv = (r > 0.f) ? 1.f / r : 0.f;
    float sn  = sim * inv;                    // sin(angle)
    float cs  = (r > 0.f) ? sre * inv : 1.f;  // cos(angle); angle(0)=0
    float mg  = log1pf(r);
    bf16_t* frow = feats + (size_t)t * KMIX;
    frow[ch]           = (bf16_t)mg;
    frow[NCH + ch]     = (bf16_t)sn;
    frow[2 * NCH + ch] = (bf16_t)cs;
  }
}

// ---------------------------------------------------------------------------
// z = feats @ W_mix.T + x @ W_skip.T + (bias_mix + bias_skip).
// Block = 4 waves = 64Mx64N. B staged to LDS in operand order (async copies,
// double buffered), shared by all 4 waves; A panels software-pipelined in
// registers. Two K segments (3072 over feats, 1024 over x) accumulate into
// the same f32 WMMA accumulators.
// ---------------------------------------------------------------------------
__global__ __launch_bounds__(32 * ZWAVES) void sffm_z_kernel(
    const bf16_t* __restrict__ feats, const bf16_t* __restrict__ xb,
    const bf16_t* __restrict__ wmix, const bf16_t* __restrict__ wskip,
    const float* __restrict__ bmix, const float* __restrict__ bskip,
    float* __restrict__ z) {
  __shared__ __align__(16) char smem[2 * SBUF];   // 12 KB, double buffered

  const int tid  = threadIdx.x;
  const int lane = tid & 31;
  const int wave = tid >> 5;          // wave w: m-subtile w AND stages ntile w
  const int half = lane >> 4;
  const int l15  = lane & 15;
  const int n0   = blockIdx.x * 64;               // x = n-group (L2 locality)
  const int m0   = blockIdx.y * (16 * ZWAVES) + wave * 16;

  v8f acc[4] = {};

  // Stage one K-step of B (4 ntiles x 16 n x 32 k) into LDS in operand order:
  // slot (ntile*32+lane) holds W[n0+ntile*16+(lane&15)][k0+(lane>>4)*16 .. +15]
  auto stage = [&](int buf, int k0, const bf16_t* w, int kw) {
    const bf16_t* g =
        w + (size_t)(n0 + wave * 16 + l15) * kw + k0 + half * 16;
    sffm_cp32(g, smem + buf * SBUF + (wave * 32 + lane) * BSTRIDE);
  };

  auto run_seg = [&](const bf16_t* amat, const bf16_t* w, int kw) {
    __syncthreads();                   // previous buffer reads done
    stage(0, 0, w, kw);
    const bf16_t* arow = amat + (size_t)(m0 + l15) * kw;
    v16bf acur = ld16x2(arow + half * 16, arow + half * 16 + 8);
    const int nsteps = kw / 32;
    for (int i = 0; i < nsteps; ++i) {
      sffm_wait_async();
      __syncthreads();                 // stage i visible to all waves
      if (i + 1 < nsteps) stage((i + 1) & 1, i * 32 + 32, w, kw);
      v16bf anext = acur;
      if (i + 1 < nsteps)
        anext = ld16x2(arow + i * 32 + 32 + half * 16,
                       arow + i * 32 + 40 + half * 16);
      const char* bb = smem + (i & 1) * SBUF;
#pragma unroll
      for (int nt = 0; nt < 4; ++nt) {
        const char* p = bb + (nt * 32 + lane) * BSTRIDE;
        v16bf b = ld16x2((const bf16_t*)p, (const bf16_t*)(p + 16));
        acc[nt] = __builtin_amdgcn_wmma_f32_16x16x32_bf16(
            false, acur, false, b, (short)0, acc[nt], false, false);
      }
      acur = anext;
    }
  };

  run_seg(feats, wmix, KMIX);   // z += feats @ W_mix.T
  run_seg(xb, wskip, IN_DIM);   // z += x @ W_skip.T

#pragma unroll
  for (int nt = 0; nt < 4; ++nt) {
    int col = n0 + nt * 16 + l15;
    float bvv = bmix[col] + bskip[col];
#pragma unroll
    for (int i = 0; i < 8; ++i) {
      int row = m0 + i + half * 8;
      z[(size_t)row * OUT_DIM + col] = acc[nt][i] + bvv;
    }
  }
}

// ---------------------------------------------------------------------------
extern "C" void kernel_launch(void* const* d_in, const int* in_sizes, int n_in,
                              void* d_out, int out_size, void* d_ws,
                              size_t ws_size, hipStream_t stream) {
  const float* x         = (const float*)d_in[0];
  const float* state     = (const float*)d_in[1];
  const int*   start     = (const int*)d_in[2];
  /* d_in[3] = next_done, unused by reference */
  const float* a_param   = (const float*)d_in[4];
  const float* b_param   = (const float*)d_in[5];
  const float* W_pre     = (const float*)d_in[6];
  const float* bias_pre  = (const float*)d_in[7];
  const float* W_skip    = (const float*)d_in[8];
  const float* bias_skip = (const float*)d_in[9];
  const float* W_mix     = (const float*)d_in[10];
  const float* bias_mix  = (const float*)d_in[11];
  (void)in_sizes; (void)n_in; (void)out_size;

  // Workspace carve-up (all region sizes are multiples of 256 bytes).
  char* ws = (char*)d_ws;
  bf16_t* xb      = (bf16_t*)ws;  ws += (size_t)T_DIM * IN_DIM * 2;    // 32 MB
  bf16_t* wpreb   = (bf16_t*)ws;  ws += (size_t)TRACE * IN_DIM * 2;
  bf16_t* wskipb  = (bf16_t*)ws;  ws += (size_t)OUT_DIM * IN_DIM * 2;
  bf16_t* wmixb   = (bf16_t*)ws;  ws += (size_t)OUT_DIM * KMIX * 2;    // 6 MB
  float*  pre     = (float*)ws;   ws += (size_t)T_DIM * TRACE * 4;     // 4 MB
  bf16_t* feats   = (bf16_t*)ws;  ws += (size_t)T_DIM * KMIX * 2;      // 96 MB
  float4* chunkAB = (float4*)ws;  ws += (size_t)NCHUNK * NCH * 16;     // 4 MB
  float2* carry   = (float2*)ws;  ws += (size_t)NCHUNK * NCH * 8;      // 2 MB
  (void)ws_size;

  // bf16 staging of activations/weights.
  {
    int n4;
    n4 = (T_DIM * IN_DIM) / 4;
    sffm_cvt_bf16<<<(n4 + 255) / 256, 256, 0, stream>>>(x, xb, n4);
    n4 = (TRACE * IN_DIM) / 4;
    sffm_cvt_bf16<<<(n4 + 255) / 256, 256, 0, stream>>>(W_pre, wpreb, n4);
    n4 = (OUT_DIM * IN_DIM) / 4;
    sffm_cvt_bf16<<<(n4 + 255) / 256, 256, 0, stream>>>(W_skip, wskipb, n4);
    n4 = (OUT_DIM * KMIX) / 4;
    sffm_cvt_bf16<<<(n4 + 255) / 256, 256, 0, stream>>>(W_mix, wmixb, n4);
  }

  // 1) pre-projection + row norm (WMMA).
  sffm_pre_kernel<<<T_DIM / 16, 32, 0, stream>>>(xb, wpreb, bias_pre, pre);

  // 2) chunked associative scan.
  float* zout   = (float*)d_out;
  float* fstate = zout + (size_t)T_DIM * OUT_DIM;  // complex64 interleaved
  sffm_scan_phaseA<<<dim3(NCHUNK, NCH / 256), 256, 0, stream>>>(
      pre, start, a_param, b_param, chunkAB);
  sffm_scan_phaseB<<<1, NCH, 0, stream>>>(chunkAB, state, carry, fstate);
  sffm_scan_phaseC<<<dim3(NCHUNK, NCH / 256), 256, 0, stream>>>(
      pre, start, a_param, b_param, carry, feats);

  // 3) fused output GEMMs (WMMA, async-LDS-staged B).
  sffm_z_kernel<<<dim3(OUT_DIM / 64, T_DIM / (16 * ZWAVES)), 32 * ZWAVES, 0,
                  stream>>>(feats, xb, wmixb, wskipb, bias_mix, bias_skip,
                            zout);
}
